// AttentionLayer_27831388078903
// MI455X (gfx1250) — matrix operand order repaired and checked by reading, not compiled, not run on hardware
//
#include <hip/hip_runtime.h>
#include <hip/hip_bf16.h>

typedef __attribute__((ext_vector_type(16))) __bf16 v16bf;
typedef __attribute__((ext_vector_type(8)))  __bf16 v8bf;
typedef __attribute__((ext_vector_type(2)))  __bf16 v2bf;
typedef __attribute__((ext_vector_type(8)))  float  v8f;

#define H 1024
#define S 2048
#define BATCH 32
#define LDSSTR 1032   // 1024 + 8 bf16 pad -> consecutive rows shift 4 LDS banks

// ---- hardware tanh (V_TANH_F32 on gfx1250) with safe fallback ----
__device__ __forceinline__ float fast_tanh(float x) {
#if __has_builtin(__builtin_amdgcn_tanhf)
    return __builtin_amdgcn_tanhf(x);
#else
    return tanhf(x);
#endif
}

// ---- f32 pair -> packed bf16 (v_cvt_pk_bf16_f32 when available) ----
__device__ __forceinline__ unsigned bf16pack(float a, float b) {
#if __has_builtin(__builtin_amdgcn_cvt_pk_bf16_f32)
    union { v2bf v; unsigned u; } p;
    p.v = __builtin_amdgcn_cvt_pk_bf16_f32(a, b);
    return p.u;
#else
    unsigned ua = __float_as_uint(a), ub = __float_as_uint(b);
    ua = (ua + 0x7FFFu + ((ua >> 16) & 1u)) >> 16;
    ub = (ub + 0x7FFFu + ((ub >> 16) & 1u)) >> 16;
    return ua | (ub << 16);
#endif
}

// ---------------------------------------------------------------------------
// Kernel 0: convert Ua_w [H][H] f32 -> bf16 (packed u32) into workspace
// ---------------------------------------------------------------------------
__global__ void cvt_ua_kernel(const float* __restrict__ Ua, unsigned* __restrict__ out) {
    int i = blockIdx.x * blockDim.x + threadIdx.x;   // one u32 = 2 elements
    if (i < H * H / 2) {
        float2 v = ((const float2*)Ua)[i];
        out[i] = bf16pack(v.x, v.y);
    }
}

// ---------------------------------------------------------------------------
// Kernel 1: qbias[b][o] = query[b,:] . Wa_w[o,:] + Wa_b[o] + Ua_b[o]
// ---------------------------------------------------------------------------
__global__ __launch_bounds__(256)
void qbias_kernel(const float* __restrict__ q,   const float* __restrict__ Wa,
                  const float* __restrict__ Wab, const float* __restrict__ Uab,
                  float* __restrict__ qbias) {
    const int lane = threadIdx.x & 31;
    const int o    = blockIdx.x * 8 + (threadIdx.x >> 5);
    float w[32];
#pragma unroll
    for (int j = 0; j < 32; ++j) w[j] = Wa[o * H + j * 32 + lane];
    const float bias = Wab[o] + Uab[o];
    for (int b = 0; b < BATCH; ++b) {
        float acc = 0.f;
#pragma unroll
        for (int j = 0; j < 32; ++j) acc += w[j] * q[b * H + j * 32 + lane];
        acc += __shfl_xor(acc, 16);
        acc += __shfl_xor(acc, 8);
        acc += __shfl_xor(acc, 4);
        acc += __shfl_xor(acc, 2);
        acc += __shfl_xor(acc, 1);
        if (lane == 0) qbias[b * H + o] = acc + bias;
    }
}

// ---------------------------------------------------------------------------
// Kernel 2: fused  scores[m] = Va . tanh(keys[m,:] @ Ua^T + qbias[b,:])
// WG = 256 threads (8 waves). M-tile = 128 rows staged to LDS as bf16.
// Waves 0-3 own row-tiles for nt 0..31, waves 4-7 for nt 32..63 (score is
// additive over n). Each wave: two 16-row sub-tiles share one B fragment per
// K-step -> 2x v_wmma_f32_16x16x32_bf16 per 32B B load; 2 waves/SIMD.
// ---------------------------------------------------------------------------
__global__ __launch_bounds__(256)
void bahdanau_scores_kernel(const float*    __restrict__ keys,
                            const unsigned* __restrict__ Ub_u32,   // Ua bf16 [H][H]
                            const float*    __restrict__ qbias,    // [B][H]
                            const float*    __restrict__ Va,       // [H]
                            float*          __restrict__ scores)   // [B*S]
{
    extern __shared__ char smem[];
    __bf16*   As   = (__bf16*)smem;        // [128][LDSSTR]
    unsigned* As_u = (unsigned*)smem;

    const int tid = threadIdx.x;
    const int m0  = blockIdx.x * 128;
    const int b   = m0 / S;                 // 128 | S, so whole tile is one batch

    // ---- stage keys tile [128 x H] f32 -> bf16 LDS ----
    const float4* k4 = (const float4*)(keys + (size_t)m0 * H);
    for (int i = tid; i < 128 * (H / 4); i += 256) {
        int row = i >> 8;                   // H/4 = 256 float4 per row
        int c4  = i & 255;
        float4 v = k4[row * 256 + c4];
        unsigned p0 = bf16pack(v.x, v.y);
        unsigned p1 = bf16pack(v.z, v.w);
        int e = row * LDSSTR + c4 * 4;      // even element offset
        As_u[(e >> 1) + 0] = p0;
        As_u[(e >> 1) + 1] = p1;
    }
    __syncthreads();

    const int wave    = tid >> 5;
    const int wgroup  = wave & 3;           // row-tile group (rows wgroup*32..+31)
    const int nhalf   = wave >> 2;          // N-range half (nt 0..31 / 32..63)
    const int lane    = tid & 31;
    const int half    = lane >> 4;          // lane group 0 / 1
    const int ln      = lane & 15;
    const int lane_ka = half * 8;           // A-frag K chunks: {0..7,16..23} vs {8..15,24..31}
    const int lane_kb = half * 16;          // unused: B-frag is loaded via lane_ka, same K chunks as A

    const __bf16* Ub    = (const __bf16*)Ub_u32;
    const __bf16* arow0 = As + (wgroup * 32 + ln)      * LDSSTR;   // sub-tile 0
    const __bf16* arow1 = As + (wgroup * 32 + 16 + ln) * LDSSTR;   // sub-tile 1

    float sacc0[8], sacc1[8];
#pragma unroll
    for (int j = 0; j < 8; ++j) { sacc0[j] = 0.f; sacc1[j] = 0.f; }

    for (int ntl = 0; ntl < 32; ++ntl) {
        const int nt = nhalf * 32 + ntl;
        const int n  = nt * 16 + ln;                // this lane's output column
        const __bf16* brow = Ub + (size_t)n * H + lane_ka;
        v8f acc0 = {}, acc1 = {};
#pragma unroll 2
        for (int k = 0; k < H; k += 32) {
            union { v16bf v; v8bf h[2]; } a0, a1, bm;
            bm.h[0] = *(const v8bf*)(brow + k);                  // 2x16B global (WGP$/L2-hot)
            bm.h[1] = *(const v8bf*)(brow + k + 16);
            a0.h[0] = *(const v8bf*)(arow0 + k + lane_ka);       // ds_load_b128
            a0.h[1] = *(const v8bf*)(arow0 + k + lane_ka + 16);
            a1.h[0] = *(const v8bf*)(arow1 + k + lane_ka);
            a1.h[1] = *(const v8bf*)(arow1 + k + lane_ka + 16);
            acc0 = __builtin_amdgcn_wmma_f32_16x16x32_bf16(false, a0.v, false, bm.v,
                                                           (short)0, acc0, false, false);
            acc1 = __builtin_amdgcn_wmma_f32_16x16x32_bf16(false, a1.v, false, bm.v,
                                                           (short)0, acc1, false, false);
        }
        const float qv = qbias[b * H + n];
        const float va = Va[n];
#pragma unroll
        for (int j = 0; j < 8; ++j) {
            sacc0[j] += fast_tanh(acc0[j] + qv) * va;
            sacc1[j] += fast_tanh(acc1[j] + qv) * va;
        }
    }

    // reduce over the 16 N-columns held across lanes of each half
#pragma unroll
    for (int j = 0; j < 8; ++j) {
        float v0 = sacc0[j], v1 = sacc1[j];
        v0 += __shfl_xor(v0, 1); v1 += __shfl_xor(v1, 1);
        v0 += __shfl_xor(v0, 2); v1 += __shfl_xor(v1, 2);
        v0 += __shfl_xor(v0, 4); v1 += __shfl_xor(v1, 4);
        v0 += __shfl_xor(v0, 8); v1 += __shfl_xor(v1, 8);
        sacc0[j] = v0; sacc1[j] = v1;
    }

    // cross-wave (N-half) combine through LDS (A tile no longer needed)
    __syncthreads();
    float* part = (float*)smem;             // [2][128]
    if (ln == 0) {
        const int base = wgroup * 32 + half * 8;   // C layout: vgpr j -> row j (+8 high half)
#pragma unroll
        for (int j = 0; j < 8; ++j) {
            part[nhalf * 128 + base + j]      = sacc0[j];
            part[nhalf * 128 + base + 16 + j] = sacc1[j];
        }
    }
    __syncthreads();
    if (tid < 128) scores[m0 + tid] = part[tid] + part[128 + tid];
}

// ---------------------------------------------------------------------------
// Kernel 3: softmax over S per batch row (Va_b omitted: shift-invariant)
// ---------------------------------------------------------------------------
__global__ __launch_bounds__(256)
void softmax_kernel(const float* __restrict__ scores, float* __restrict__ w_out) {
    __shared__ float red[256];
    const int b = blockIdx.x, tid = threadIdx.x;
    const float* s = scores + b * S;
    float v[8];
    float mx = -1e30f;
#pragma unroll
    for (int j = 0; j < 8; ++j) { v[j] = s[tid + j * 256]; mx = fmaxf(mx, v[j]); }
    red[tid] = mx; __syncthreads();
    for (int off = 128; off; off >>= 1) {
        if (tid < off) red[tid] = fmaxf(red[tid], red[tid + off]);
        __syncthreads();
    }
    mx = red[0]; __syncthreads();
    float sum = 0.f;
#pragma unroll
    for (int j = 0; j < 8; ++j) { v[j] = __expf(v[j] - mx); sum += v[j]; }
    red[tid] = sum; __syncthreads();
    for (int off = 128; off; off >>= 1) {
        if (tid < off) red[tid] += red[tid + off];
        __syncthreads();
    }
    const float inv = 1.f / red[0];
#pragma unroll
    for (int j = 0; j < 8; ++j) w_out[b * S + tid + j * 256] = v[j] * inv;
}

// ---------------------------------------------------------------------------
// Kernel 4a: partial context over an s-chunk:
//   part[b][c][h] = sum_{s in chunk c} weights[b,s] * keys[b,s,h]
// grid = B x 8 chunks; thread handles 4 h via float4 (fully coalesced).
// ---------------------------------------------------------------------------
#define SCHUNKS 8
__global__ __launch_bounds__(256)
void context_part_kernel(const float* __restrict__ w, const float* __restrict__ keys,
                         float* __restrict__ part) {
    const int b  = blockIdx.x >> 3;
    const int c  = blockIdx.x & 7;
    const int s0 = c * (S / SCHUNKS);
    const float4* kb = (const float4*)(keys + (size_t)b * S * H) + (size_t)s0 * (H / 4);
    const float*  wb = w + b * S + s0;
    float4 acc = make_float4(0.f, 0.f, 0.f, 0.f);
#pragma unroll 4
    for (int s = 0; s < S / SCHUNKS; ++s) {
        float ws = wb[s];
        float4 k = kb[(size_t)s * (H / 4) + threadIdx.x];
        acc.x += ws * k.x; acc.y += ws * k.y; acc.z += ws * k.z; acc.w += ws * k.w;
    }
    ((float4*)part)[((size_t)b * SCHUNKS + c) * (H / 4) + threadIdx.x] = acc;
}

// Kernel 4b: deterministic reduction of the 8 partials -> context[b][h]
__global__ __launch_bounds__(256)
void context_reduce_kernel(const float* __restrict__ part, float* __restrict__ ctx) {
    const int i = blockIdx.x * 256 + threadIdx.x;   // i = b*H + h, BATCH*H total
    const int b = i >> 10;
    const int h = i & 1023;
    float acc = 0.f;
#pragma unroll
    for (int c = 0; c < SCHUNKS; ++c)
        acc += part[((size_t)b * SCHUNKS + c) * H + h];
    ctx[i] = acc;
}

// ---------------------------------------------------------------------------
extern "C" void kernel_launch(void* const* d_in, const int* in_sizes, int n_in,
                              void* d_out, int out_size, void* d_ws, size_t ws_size,
                              hipStream_t stream) {
    const float* query = (const float*)d_in[0];
    const float* keys  = (const float*)d_in[1];
    const float* Wa_w  = (const float*)d_in[2];
    const float* Wa_b  = (const float*)d_in[3];
    const float* Ua_w  = (const float*)d_in[4];
    const float* Ua_b  = (const float*)d_in[5];
    const float* Va_w  = (const float*)d_in[6];
    // d_in[7] = Va_b: unused, softmax is invariant to a uniform score shift

    char* ws = (char*)d_ws;
    unsigned* Ub_u32 = (unsigned*)ws;                               // 2 MB bf16 Ua
    size_t off = (size_t)H * H * 2;
    float* qbias  = (float*)(ws + off); off += (size_t)BATCH * H * 4;     // 128 KB
    float* scores = (float*)(ws + off); off += (size_t)BATCH * S * 4;     // 256 KB
    float* cpart  = (float*)(ws + off);                                   // 1 MB

    float* ctx_out = (float*)d_out;               // context [B][H]
    float* w_out   = (float*)d_out + BATCH * H;   // weights [B][S]

    cvt_ua_kernel<<<(H * H / 2) / 256, 256, 0, stream>>>(Ua_w, Ub_u32);
    qbias_kernel<<<H / 8, 256, 0, stream>>>(query, Wa_w, Wa_b, Ua_b, qbias);

    const size_t lds_bytes = 128 * LDSSTR * sizeof(unsigned short);     // ~258 KB
    bahdanau_scores_kernel<<<(BATCH * S) / 128, 256, lds_bytes, stream>>>(
        keys, Ub_u32, qbias, Va_w, scores);

    softmax_kernel<<<BATCH, 256, 0, stream>>>(scores, w_out);
    context_part_kernel<<<BATCH * SCHUNKS, 256, 0, stream>>>(w_out, keys, cpart);
    context_reduce_kernel<<<BATCH * H / 256, 256, 0, stream>>>(cpart, ctx_out);
}
